// Generator_28845000360693
// MI455X (gfx1250) — matrix operand order repaired and checked by reading, not compiled, not run on hardware
//
#include <hip/hip_runtime.h>

// ---------- types ----------
typedef __attribute__((ext_vector_type(16))) __bf16 v16bf;
typedef __attribute__((ext_vector_type(8)))  float  v8f;
typedef __attribute__((ext_vector_type(8)))  unsigned short ushort8;

union BF16x16 {
    struct { ushort8 lo, hi; } u;
    v16bf v;
};

__device__ __forceinline__ unsigned short f32_to_bf16(float f) {
    union { float f; unsigned int u; } v; v.f = f;
    unsigned int u = v.u;
    u += 0x7fffu + ((u >> 16) & 1u);   // round to nearest even
    return (unsigned short)(u >> 16);
}

// ---------- constants ----------
#define NPOS   8192      // B * 32 * 32
#define KDIM   1536
#define ODIM   1536

// ---------- kernel 1: W fp32 -> bf16 ----------
__global__ __launch_bounds__(256) void wconv_kernel(const float* __restrict__ W,
                                                    unsigned short* __restrict__ Wb) {
    int i = (blockIdx.x * 256 + threadIdx.x) * 4;     // 2304 blocks * 256 * 4 = 2359296
    float4 v = *reinterpret_cast<const float4*>(W + i);
    unsigned long long p =
        (unsigned long long)f32_to_bf16(v.x)
      | ((unsigned long long)f32_to_bf16(v.y) << 16)
      | ((unsigned long long)f32_to_bf16(v.z) << 32)
      | ((unsigned long long)f32_to_bf16(v.w) << 48);
    *reinterpret_cast<unsigned long long*>(Wb + i) = p;
}

// ---------- kernel 2: feat0 (mean of 6 consecutive flattened patch elems of x0) ----------
__global__ __launch_bounds__(256) void feat0_kernel(const float* __restrict__ x0,
                                                    unsigned short* __restrict__ X) {
    int idx = blockIdx.x * 256 + threadIdx.x;         // 24576 * 256 = 8192*768 exact
    int n = idx / 768;
    int c = idx - n * 768;
    int b = n >> 10, hw = n & 1023, h = hw >> 5, w = hw & 31;
    int k = c / 3, r = c - 3 * k;
    const float* base = x0 + (size_t)b * 512 * 1024;

    float s = 0.f;
    auto tap = [&](int ch, int ki, int kj) -> float {
        int hh = h + ki - 1, ww = w + kj - 1;
        if ((unsigned)hh < 32u && (unsigned)ww < 32u)
            return base[(size_t)ch * 1024 + hh * 32 + ww];
        return 0.f;
    };
    if (r == 0) {                // channel 2k, rows ki=0,1
        int ch = 2 * k;
        for (int ki = 0; ki < 2; ++ki)
            for (int kj = 0; kj < 3; ++kj) s += tap(ch, ki, kj);
    } else if (r == 1) {         // channel 2k row ki=2  +  channel 2k+1 row ki=0
        int ch = 2 * k;
        for (int kj = 0; kj < 3; ++kj) { s += tap(ch, 2, kj); s += tap(ch + 1, 0, kj); }
    } else {                     // channel 2k+1, rows ki=1,2
        int ch = 2 * k + 1;
        for (int ki = 1; ki < 3; ++ki)
            for (int kj = 0; kj < 3; ++kj) s += tap(ch, ki, kj);
    }
    X[(size_t)n * KDIM + c] = f32_to_bf16(s * (1.0f / 6.0f));
}

// ---------- kernel 3: feat1 pooled at 16x16 (mean of 12 flattened patch elems of x1) ----------
__global__ __launch_bounds__(256) void feat1_kernel(const float* __restrict__ x1,
                                                    float* __restrict__ F) {
    int idx = blockIdx.x * 256 + threadIdx.x;         // 6144 * 256 = 8*256*768 exact
    int pos = idx / 768;
    int cc = idx - pos * 768;
    int b = pos >> 8, yx = pos & 255, y = yx >> 4, x = yx & 15;
    const float* base = x1 + (size_t)b * 1024 * 256;

    int r  = cc % 3;             // start kernel-row within channel (kk0 = 3r)
    int ch = (cc * 4) / 3;       // start channel (= 12cc / 9)
    int ki = r, kj = 0;
    float s = 0.f;
    #pragma unroll
    for (int t = 0; t < 12; ++t) {
        int hh = y + ki - 1, ww = x + kj - 1;
        if ((unsigned)hh < 16u && (unsigned)ww < 16u)
            s += base[(size_t)ch * 256 + hh * 16 + ww];
        if (++kj == 3) { kj = 0; if (++ki == 3) { ki = 0; ++ch; } }
    }
    F[(size_t)pos * 768 + cc] = s * (1.0f / 12.0f);
}

// ---------- kernel 4: bilinear 16->32 upsample of pooled feat1 (align_corners=False) ----------
__global__ __launch_bounds__(256) void upsample_kernel(const float* __restrict__ F,
                                                       unsigned short* __restrict__ X) {
    int idx = blockIdx.x * 256 + threadIdx.x;         // 24576 * 256 = 8192*768 exact
    int n = idx / 768;
    int cc = idx - n * 768;
    int b = n >> 10, hw = n & 1023, h = hw >> 5, w = hw & 31;

    int th = h >> 1, tw = w >> 1;
    int y0, y1, xx0, xx1; float wy0, wy1, wx0, wx1;
    if (h & 1) { y0 = th; y1 = (th < 15) ? th + 1 : 15; wy0 = 0.75f; wy1 = 0.25f; }
    else       { y0 = (th > 0) ? th - 1 : 0; y1 = th;   wy0 = 0.25f; wy1 = 0.75f; }
    if (w & 1) { xx0 = tw; xx1 = (tw < 15) ? tw + 1 : 15; wx0 = 0.75f; wx1 = 0.25f; }
    else       { xx0 = (tw > 0) ? tw - 1 : 0; xx1 = tw;   wx0 = 0.25f; wx1 = 0.75f; }

    const float* Fb = F + (size_t)b * 256 * 768 + cc;
    float v = wy0 * (wx0 * Fb[(size_t)(y0 * 16 + xx0) * 768] + wx1 * Fb[(size_t)(y0 * 16 + xx1) * 768])
            + wy1 * (wx0 * Fb[(size_t)(y1 * 16 + xx0) * 768] + wx1 * Fb[(size_t)(y1 * 16 + xx1) * 768]);
    X[(size_t)n * KDIM + 768 + cc] = f32_to_bf16(v);
}

// ---------- kernel 5: WMMA bf16 GEMM with async-to-LDS double-buffered staging ----------
// out[8192,1536] = X @ Wb^T + bias.
// Block: 256 threads = 8 waves (4M x 2N), macro-tile 128(M) x 64(N), K staged 32/iter.
// LDS per stage: A 128x64B = 8 KB, B 64x64B = 4 KB -> 24 KB double-buffered.
// Grid: (8192/128) * (1536/64) = 64 * 24 = 1536 blocks.

__device__ __forceinline__ void async_ld_b128(unsigned lds_byte_off,
                                              const unsigned short* g,
                                              const unsigned short* lds_dummy) {
    // GLOBAL_LOAD_ASYNC_TO_LDS_B128: VDST = LDS byte offset VGPR, VADDR = 64-bit global addr.
    // lds_dummy makes the shared array's address escape so the "memory" clobber covers it.
    asm volatile("global_load_async_to_lds_b128 %0, %1, off"
                 :: "v"(lds_byte_off), "v"(g), "v"(lds_dummy)
                 : "memory");
}

__global__ __launch_bounds__(256) void gemm_wmma_bf16(const unsigned short* __restrict__ Xbf,
                                                      const unsigned short* __restrict__ Wb,
                                                      const float* __restrict__ bias,
                                                      float* __restrict__ out) {
    __shared__ __align__(64) unsigned short sh[12288];   // 24576 B = 2 x (8KB A + 4KB B)

    const int tid    = threadIdx.x;
    const int lane   = tid & 31;
    const int wave   = tid >> 5;
    const int blockN = blockIdx.x % 24;
    const int blockM = blockIdx.x / 24;
    const int wm     = wave >> 1;             // 0..3
    const int wn     = wave & 1;              // 0..1
    const int mbase0 = blockM * 128;
    const int nbase0 = blockN * 64;
    const int mbase  = mbase0 + wm * 32;
    const int nbase  = nbase0 + wn * 32;

    const int half = lane >> 4;               // 0 or 1 (lane group)
    const int l16  = lane & 15;

    // ---- per-thread async staging: 3 x 16B chunks per stage ----
    const int rowq = tid >> 2;                // 0..63
    const int qq   = tid & 3;                 // 16B chunk within a 64B row
    const unsigned short* gA0 = Xbf + (size_t)(mbase0 + rowq)      * KDIM + qq * 8;
    const unsigned short* gA1 = Xbf + (size_t)(mbase0 + 64 + rowq) * KDIM + qq * 8;
    const unsigned short* gB  = Wb  + (size_t)(nbase0 + rowq)      * KDIM + qq * 8;
    const unsigned ldsA0 =        rowq * 64 + qq * 16;   // bytes within stage buffer
    const unsigned ldsA1 = 4096u + rowq * 64 + qq * 16;
    const unsigned ldsB  = 8192u + rowq * 64 + qq * 16;

    // ---- per-wave WMMA fragment LDS byte offsets (ISA 16-bit A/B layouts) ----
    const unsigned fa0 = (unsigned)(wm * 32 + l16) * 64 + half * 16;          // + hi at +32
    const unsigned fa1 = fa0 + 16u * 64;
    const unsigned fb0 = 8192u + (unsigned)(wn * 32 + l16) * 64 + half * 16;  // + hi at +32
    const unsigned fb1 = fb0 + 16u * 64;

    v8f acc00 = {}, acc01 = {}, acc10 = {}, acc11 = {};

    // prime stage 0 into buffer 0
    async_ld_b128(ldsA0, gA0, sh);
    async_ld_b128(ldsA1, gA1, sh);
    async_ld_b128(ldsB,  gB,  sh);

    for (int s = 0; s < 48; ++s) {
        const unsigned pb = (unsigned)(s & 1) * 12288u;     // current stage buffer (bytes)
        if (s + 1 < 48) {
            const size_t   ke  = (size_t)(s + 1) * 32;      // element offset of next K chunk
            const unsigned pb2 = (unsigned)((s + 1) & 1) * 12288u;
            async_ld_b128(pb2 + ldsA0, gA0 + ke, sh);
            async_ld_b128(pb2 + ldsA1, gA1 + ke, sh);
            async_ld_b128(pb2 + ldsB,  gB  + ke, sh);
            asm volatile("s_wait_asynccnt 0x3" ::: "memory");   // stage s complete (in-order)
        } else {
            asm volatile("s_wait_asynccnt 0x0" ::: "memory");
        }
        __syncthreads();    // stage-s data visible to all waves

        const char* shb = reinterpret_cast<const char*>(sh);
        BF16x16 a0, a1, b0, b1;
        a0.u.lo = *reinterpret_cast<const ushort8*>(shb + pb + fa0);
        a0.u.hi = *reinterpret_cast<const ushort8*>(shb + pb + fa0 + 32);
        a1.u.lo = *reinterpret_cast<const ushort8*>(shb + pb + fa1);
        a1.u.hi = *reinterpret_cast<const ushort8*>(shb + pb + fa1 + 32);
        b0.u.lo = *reinterpret_cast<const ushort8*>(shb + pb + fb0);
        b0.u.hi = *reinterpret_cast<const ushort8*>(shb + pb + fb0 + 32);
        b1.u.lo = *reinterpret_cast<const ushort8*>(shb + pb + fb1);
        b1.u.hi = *reinterpret_cast<const ushort8*>(shb + pb + fb1 + 32);

        acc00 = __builtin_amdgcn_wmma_f32_16x16x32_bf16(false, a0.v, false, b0.v, (short)0, acc00, false, false);
        acc01 = __builtin_amdgcn_wmma_f32_16x16x32_bf16(false, a0.v, false, b1.v, (short)0, acc01, false, false);
        acc10 = __builtin_amdgcn_wmma_f32_16x16x32_bf16(false, a1.v, false, b0.v, (short)0, acc10, false, false);
        acc11 = __builtin_amdgcn_wmma_f32_16x16x32_bf16(false, a1.v, false, b1.v, (short)0, acc11, false, false);

        __syncthreads();    // all reads of buffer pb done before it is refilled (stage s+2)
    }

    const float bn0 = bias[nbase + l16];
    const float bn1 = bias[nbase + 16 + l16];
    const int   mro = mbase + half * 8;       // D layout: VGPR r -> M = r + half*8, N = l16
    #pragma unroll
    for (int r = 0; r < 8; ++r) {
        out[(size_t)(mro + r)      * ODIM + nbase + l16]      = acc00[r] + bn0;
        out[(size_t)(mro + r)      * ODIM + nbase + 16 + l16] = acc01[r] + bn1;
        out[(size_t)(mro + 16 + r) * ODIM + nbase + l16]      = acc10[r] + bn0;
        out[(size_t)(mro + 16 + r) * ODIM + nbase + 16 + l16] = acc11[r] + bn1;
    }
}

// ---------- launch ----------
extern "C" void kernel_launch(void* const* d_in, const int* in_sizes, int n_in,
                              void* d_out, int out_size, void* d_ws, size_t ws_size,
                              hipStream_t stream) {
    (void)in_sizes; (void)n_in; (void)out_size; (void)ws_size;
    const float* x0 = (const float*)d_in[0];   // [8,512,32,32]
    const float* x1 = (const float*)d_in[1];   // [8,1024,16,16]
    const float* W  = (const float*)d_in[2];   // [1536,1536]
    const float* bi = (const float*)d_in[3];   // [1536]
    float* out = (float*)d_out;                // [8192,1536]

    char* ws = (char*)d_ws;
    unsigned short* X  = (unsigned short*)(ws);               // bf16 [8192][1536]  (25165824 B)
    unsigned short* Wb = (unsigned short*)(ws + 25165824);    // bf16 [1536][1536]  ( 4718592 B)
    float*          F1 = (float*)(ws + 29884416);             // f32  [8*256][768]  ( 6291456 B)

    wconv_kernel   <<<2304,  256, 0, stream>>>(W, Wb);
    feat0_kernel   <<<24576, 256, 0, stream>>>(x0, X);
    feat1_kernel   <<<6144,  256, 0, stream>>>(x1, F1);
    upsample_kernel<<<24576, 256, 0, stream>>>(F1, X);
    gemm_wmma_bf16 <<<1536,  256, 0, stream>>>(X, Wb, bi, out);
}
